// PReNet_68693706932303
// MI455X (gfx1250) — compile-verified, matrix operand order checked by reading, not run on hardware
//
#include <hip/hip_runtime.h>
#include <hip/hip_bf16.h>
#include <math.h>

typedef __attribute__((ext_vector_type(2)))  _Float16 hf2;
typedef __attribute__((ext_vector_type(8)))  _Float16 hf8;
typedef __attribute__((ext_vector_type(16))) _Float16 v16h;
typedef __attribute__((ext_vector_type(8)))  float    v8f;

// ---------------------------------------------------------------------------
// BN-fold: wh[co][kk] = w[co][kk] * g/sqrt(v+eps)  (f16, zero-padded to Kpad)
//          bias[co]   = b*inv + be - m*inv
// ---------------------------------------------------------------------------
__global__ void k_fold_bn(const float* __restrict__ w, const float* __restrict__ b,
                          const float* __restrict__ g, const float* __restrict__ be,
                          const float* __restrict__ m, const float* __restrict__ v,
                          _Float16* __restrict__ wh, float* __restrict__ bias,
                          int Cout, int K, int Kpad) {
  long i = (long)blockIdx.x * blockDim.x + threadIdx.x;
  long total = (long)Cout * Kpad;
  if (i >= total) return;
  int co = (int)(i / Kpad), kk = (int)(i - (long)co * Kpad);
  float inv = rsqrtf(v[co] + 1e-5f) * g[co];
  wh[(long)co * Kpad + kk] = (_Float16)(kk < K ? w[(long)co * K + kk] * inv : 0.f);
  if (kk == 0) bias[co] = b[co] * inv + be[co] - m[co] * inv;
}

// fp32 -> f16 (row-major [O][K] -> [O][Kpad], zero padded)
__global__ void k_cvt_w(const float* __restrict__ w, _Float16* __restrict__ wh,
                        int O, int K, int Kpad) {
  long i = (long)blockIdx.x * blockDim.x + threadIdx.x;
  long total = (long)O * Kpad;
  if (i >= total) return;
  int o = (int)(i / Kpad), kk = (int)(i - (long)o * Kpad);
  wh[i] = (_Float16)(kk < K ? w[(long)o * K + kk] : 0.f);
}

__global__ void k_f2h(const float* __restrict__ x, _Float16* __restrict__ y, long n) {
  long i = (long)blockIdx.x * blockDim.x + threadIdx.x;
  if (i < n) y[i] = (_Float16)x[i];
}

__global__ void k_zero(float* __restrict__ p, int n) {
  int i = blockIdx.x * blockDim.x + threadIdx.x;
  if (i < n) p[i] = 0.f;
}

// ---------------------------------------------------------------------------
// Implicit-GEMM conv, B-tile shared across the block.
//   block = 256 threads = 8 waves; one 16-position column tile per block.
//   All 256 threads cooperatively stage the 16x32 f16 im2col tile (2 halfs
//   per thread, one ds_store_b32 each); each wave then multiplies the SAME
//   B fragment against its own 16-channel A fragment (blockIdx.x*8+wave).
//   A-fragment loads are unconditional from a row-clamped pointer; invalid
//   rows/cols are masked only at the epilogue stores, so the inner loop has
//   no EXEC manipulation at all.
// ---------------------------------------------------------------------------
template <int KH, int KW>
__global__ __launch_bounds__(256) void k_conv(
    const float* __restrict__ x, const _Float16* __restrict__ wh,
    const float* __restrict__ bias, const float* __restrict__ resid,
    float* __restrict__ out,
    int N, int Cin, int H, int W, int Cout,
    int stride, int pad, int OH, int OW, int Kpad, int relu) {
  __shared__ _Float16 lb[16][32];
  const int tid  = threadIdx.x;
  const int lane = tid & 31;
  const int widx = tid >> 5;
  const int ntile = blockIdx.y;

  constexpr int KHW = KH * KW;
  const int K    = Cin * KHW;
  const int Npos = N * OH * OW;

  // ---- staging role: this thread owns column (tid>>4), K pair (tid&15)*2 ----
  const int scol = tid >> 4;
  const int kk0  = (tid & 15) * 2;
  const int sp0  = ntile * 16 + scol;
  const int sp   = (sp0 < Npos) ? sp0 : 0;   // clamp: garbage cols masked at store
  int simg, soy, sox;
  {
    simg = sp / (OH * OW);
    int rem = sp - simg * (OH * OW);
    soy = rem / OW; sox = rem - soy * OW;
  }
  const int siy0 = soy * stride - pad;
  const int six0 = sox * stride - pad;
  const float* sxb = x + (long)simg * Cin * H * W;

  // ---- compute role: lane's output column + wave's channel tile ----
  const int c16  = lane & 15;
  const int cp   = ntile * 16 + c16;
  const bool colv = (cp < Npos);
  int img, oy, ox;
  {
    int pp = colv ? cp : 0;
    img = pp / (OH * OW);
    int rem = pp - img * (OH * OW);
    oy = rem / OW; ox = rem - oy * OW;
  }
  const int kadj  = (lane >> 4) * 8;
  const int mtile = blockIdx.x * 8 + widx;
  const int mrow  = mtile * 16 + c16;
  const bool mv   = (mrow < Cout);
  const _Float16* wrow = wh + (long)(mv ? mrow : 0) * Kpad + kadj;  // clamped

  v8f acc = {};
  for (int kc = 0; kc < Kpad; kc += 32) {
    // ---- cooperative stage: 2 halfs per thread, one b32 LDS store ----
    int k0 = kc + kk0;
    float v0 = 0.f, v1 = 0.f;
    {
      int ci, r, s;
      if (KHW == 1) {
        ci = k0; r = 0; s = 0;
      } else {
        ci = k0 / KHW;
        int rr = k0 - ci * KHW;
        r = rr / KW; s = rr - r * KW;
      }
      if (k0 < K) {
        int iy = siy0 + r, ix = six0 + s;
        if ((unsigned)iy < (unsigned)H && (unsigned)ix < (unsigned)W)
          v0 = sxb[(ci * H + iy) * W + ix];
      }
      if (KHW == 1) {
        ++ci;
      } else {
        if (++s == KW) { s = 0; if (++r == KH) { r = 0; ++ci; } }
      }
      if (k0 + 1 < K) {
        int iy = siy0 + r, ix = six0 + s;
        if ((unsigned)iy < (unsigned)H && (unsigned)ix < (unsigned)W)
          v1 = sxb[(ci * H + iy) * W + ix];
      }
    }
    hf2 pk; pk[0] = (_Float16)v0; pk[1] = (_Float16)v1;
    *(hf2*)&lb[scol][kk0] = pk;
    __syncthreads();

    // ---- A fragment: two global b128 loads (unconditional, clamped row) ----
    hf8 a0 = *(const hf8*)(wrow + kc);
    hf8 a1 = *(const hf8*)(wrow + kc + 16);
    __builtin_prefetch((const void*)(wrow + kc + 32), 0, 1);
    // ---- B fragment: two ds_load_b128, shared by all 8 waves ----
    const _Float16* bp = &lb[c16][kadj];
    hf8 b0 = *(const hf8*)bp;
    hf8 b1 = *(const hf8*)(bp + 16);

    v16h av = __builtin_shufflevector(a0, a1, 0,1,2,3,4,5,6,7,8,9,10,11,12,13,14,15);
    v16h bv = __builtin_shufflevector(b0, b1, 0,1,2,3,4,5,6,7,8,9,10,11,12,13,14,15);
    acc = __builtin_amdgcn_wmma_f32_16x16x32_f16(false, av, false, bv,
                                                 (short)0, acc, false, false);
    __syncthreads();
  }

  // ---- epilogue: bias (+residual) (+relu) ----
  if (colv) {
#pragma unroll
    for (int j = 0; j < 8; ++j) {
      int co = mtile * 16 + ((lane >> 4) * 8) + j;   // D: M = j + 8*(lane>=16)
      if (co < Cout) {
        int idx = ((img * Cout + co) * OH + oy) * OW + ox;
        float v = acc[j] + bias[co];
        if (resid) v += resid[idx];
        if (relu)  v = fmaxf(v, 0.f);
        out[idx] = v;
      }
    }
  }
}

// ---------------------------------------------------------------------------
// FC GEMM: y[b][o] = sum_k xh[b][k] * wh[o][k]. One wave per 16x16 tile.
// Loads are unconditional from row/col-clamped pointers; masking happens at
// the store only. mode 0: split-K partial via global_atomic_add_f32.
// mode 1: full-K, +bias, act: 1=relu, 2=sigmoid, 3=3.14*sigmoid.
// ---------------------------------------------------------------------------
__global__ __launch_bounds__(32) void k_fc(
    const _Float16* __restrict__ xh, const _Float16* __restrict__ wh,
    const float* __restrict__ bias, float* __restrict__ out,
    int Brows, int O, int Kpad, int kSpan, int mode, int act) {
  const int lane  = threadIdx.x & 31;
  const int mtile = blockIdx.x, ntile = blockIdx.y;
  const int kstart = blockIdx.z * kSpan;
  const int kend   = min(kstart + kSpan, Kpad);

  const int  c16 = lane & 15;
  const int  row = min(mtile * 16 + c16, Brows - 1);  // clamp
  const int  col = min(ntile * 16 + c16, O - 1);      // clamp
  const int  kadj = (lane >> 4) * 8;
  const _Float16* xr = xh + (long)row * Kpad + kadj;
  const _Float16* wr = wh + (long)col * Kpad + kadj;

  v8f acc = {};
  for (int kc = kstart; kc < kend; kc += 32) {
    hf8 a0 = *(const hf8*)(xr + kc);
    hf8 a1 = *(const hf8*)(xr + kc + 16);
    hf8 b0 = *(const hf8*)(wr + kc);
    hf8 b1 = *(const hf8*)(wr + kc + 16);
    v16h av = __builtin_shufflevector(a0, a1, 0,1,2,3,4,5,6,7,8,9,10,11,12,13,14,15);
    v16h bv = __builtin_shufflevector(b0, b1, 0,1,2,3,4,5,6,7,8,9,10,11,12,13,14,15);
    acc = __builtin_amdgcn_wmma_f32_16x16x32_f16(false, av, false, bv,
                                                 (short)0, acc, false, false);
  }

#pragma unroll
  for (int j = 0; j < 8; ++j) {
    int r2 = mtile * 16 + ((lane >> 4) * 8) + j;
    int c2 = ntile * 16 + c16;
    if (r2 < Brows && c2 < O) {
      if (mode == 0) {
        atomicAdd(&out[r2 * O + c2], acc[j]);
      } else {
        float v = acc[j] + bias[c2];
        if (act == 1)      v = fmaxf(v, 0.f);
        else if (act == 2) v = 1.f / (1.f + __expf(-v));
        else if (act == 3) v = 3.14f / (1.f + __expf(-v));
        out[r2 * O + c2] = v;
      }
    }
  }
}

__global__ void k_fc_epi(const float* __restrict__ acc, const float* __restrict__ bias,
                         _Float16* __restrict__ h16, int O, int n) {
  int i = blockIdx.x * blockDim.x + threadIdx.x;
  if (i >= n) return;
  float v = fmaxf(acc[i] + bias[i % O], 0.f);
  h16[i] = (_Float16)v;
}

// ---------------------------------------------------------------------------
// 3x3 stride-2 VALID max pool
// ---------------------------------------------------------------------------
__global__ void k_maxpool(const float* __restrict__ x, float* __restrict__ y,
                          int N, int C, int H, int W, int OH, int OW) {
  int i = blockIdx.x * blockDim.x + threadIdx.x;
  int total = N * C * OH * OW;
  if (i >= total) return;
  int ox = i % OW; int t = i / OW;
  int oy = t % OH; t /= OH;
  int c  = t % C;  int n = t / C;
  const float* p = x + ((long)(n * C + c) * H + oy * 2) * W + ox * 2;
  float m = -INFINITY;
#pragma unroll
  for (int r = 0; r < 3; ++r)
#pragma unroll
    for (int s = 0; s < 3; ++s) m = fmaxf(m, p[r * W + s]);
  y[i] = m;
}

// ---------------------------------------------------------------------------
// Forward kinematics: one thread per batch element
// ---------------------------------------------------------------------------
__constant__ float c_phal[20] = {
  40.712f, 34.04f, 29.417f, 26.423f, 79.706f, 35.224f, 23.27f, 22.036f,
  75.669f, 41.975f, 26.329f, 24.504f, 75.358f, 39.978f, 23.513f, 22.647f,
  74.556f, 27.541f, 19.826f, 20.395f};

__device__ __forceinline__ void mat4mul(const float* A, const float* B, float* C) {
#pragma unroll
  for (int i = 0; i < 4; ++i)
#pragma unroll
    for (int j = 0; j < 4; ++j) {
      float s = 0.f;
#pragma unroll
      for (int t = 0; t < 4; ++t) s += A[i * 4 + t] * B[t * 4 + j];
      C[i * 4 + j] = s;
    }
}
__device__ __forceinline__ void chain(float* T, float thz, float alx, float d) {
  float c1 = cosf(thz), s1 = sinf(thz);
  float c2 = cosf(alx), s2 = sinf(alx);
  float Z[16] = {c1,-s1,0,0,  s1,c1,0,0,  0,0,1,0,  0,0,0,1};
  float X[16] = {1,0,0,d,  0,c2,-s2,0,  0,s2,c2,0,  0,0,0,1};
  float P[16], R[16];
  mat4mul(Z, X, P);
  mat4mul(T, P, R);
#pragma unroll
  for (int i = 0; i < 16; ++i) T[i] = R[i];
}

__global__ void k_fk(const float* __restrict__ scale, const float* __restrict__ theta,
                     float* __restrict__ pos, int B) {
  int b = blockIdx.x * blockDim.x + threadIdx.x;
  if (b >= B) return;
  const float* th = theta + b * 33;
  float s[20];
#pragma unroll
  for (int j = 0; j < 20; ++j) s[j] = scale[b * 20 + j] + 1.f;
  float* pb = pos + b * 63;  // 21 joints x 3
  pb[0] = 0.f; pb[1] = 0.f; pb[2] = 0.f;
  float T[16];
#pragma unroll
  for (int q = 0; q < 16; ++q) T[q] = (q % 5 == 0) ? 1.f : 0.f;
  auto store = [&](int idx) {
    pb[idx * 3 + 0] = T[3]; pb[idx * 3 + 1] = T[7]; pb[idx * 3 + 2] = T[11];
  };
  // thumb
  chain(T, th[0], th[1], 0.f);
  chain(T, th[2], th[3], s[0] * c_phal[0]); store(1);
  chain(T, th[4], th[5], 0.f);
  chain(T, th[6], 0.f, s[1] * c_phal[1]); store(2);
  chain(T, th[7], 0.f, s[2] * c_phal[2]); store(3);
  chain(T, th[8], 0.f, s[3] * c_phal[3]); store(4);
  // four fingers (T reset at each root)
  int j = 4;
  for (int k = 0; k < 4; ++k) {
#pragma unroll
    for (int q = 0; q < 16; ++q) T[q] = (q % 5 == 0) ? 1.f : 0.f;
    chain(T, th[6 * k + 9], 0.f, s[j] * c_phal[j]); ++j; store(4 * k + 5);
    chain(T, th[6 * k + 10], th[6 * k + 11], 0.f);
    for (int l = 0; l < 3; ++l) {
      chain(T, th[6 * k + 12 + l], 0.f, s[j] * c_phal[j]); ++j;
      store(4 * k + 6 + l);
    }
  }
}

// ---------------------------------------------------------------------------
// Host orchestration
// ---------------------------------------------------------------------------
extern "C" void kernel_launch(void* const* d_in, const int* in_sizes, int n_in,
                              void* d_out, int out_size, void* d_ws, size_t ws_size,
                              hipStream_t stream) {
  (void)in_sizes; (void)n_in; (void)out_size; (void)ws_size;
  const int N = 32;
  int ic = 0;
  auto in_f = [&]() { return (const float*)d_in[ic++]; };

  const float* X = in_f();

  struct CBp { const float *w, *b, *g, *be, *m, *v; };
  auto getCB = [&]() {
    CBp c; c.w = in_f(); c.b = in_f(); c.g = in_f();
    c.be = in_f(); c.m = in_f(); c.v = in_f(); return c;
  };

  // workspace layout
  const size_t ACT = 33554432;           // floats per activation buffer (32*64*128*128)
  float*     bufA = (float*)d_ws;
  float*     bufB = bufA + ACT;
  float*     bufC = bufB + ACT;
  _Float16*  WAR  = (_Float16*)(bufC + ACT);   // f16 weight arena
  const size_t WARN = 50331648;                // 48M halfs
  float*     BAR  = (float*)(WAR + WARN);      // folded biases
  _Float16*  XH   = (_Float16*)(BAR + 16384);  // flat activations f16 [32][65536]
  float*     HT   = (float*)(XH + (size_t)N * 65536);
  float*     HS   = HT + N * 256;
  _Float16*  HT16 = (_Float16*)(HS + N * 256);
  _Float16*  HS16 = HT16 + N * 256;

  size_t wcur = 0, bcur = 0;
  struct F { const _Float16* w; const float* b; int Kpad; };
  auto foldCB = [&](CBp p, int cin, int cout, int k) -> F {
    int K = cin * k * k, Kpad = ((K + 31) / 32) * 32;
    _Float16* wd = WAR + wcur; float* bd = BAR + bcur;
    long total = (long)cout * Kpad;
    k_fold_bn<<<dim3((unsigned)((total + 255) / 256)), 256, 0, stream>>>(
        p.w, p.b, p.g, p.be, p.m, p.v, wd, bd, cout, K, Kpad);
    wcur += (size_t)total; bcur += cout;
    return F{wd, bd, Kpad};
  };
  auto cvt = [&](const float* w, int O, int K) -> const _Float16* {
    int Kpad = ((K + 31) / 32) * 32;
    _Float16* d = WAR + wcur;
    long n = (long)O * Kpad;
    k_cvt_w<<<dim3((unsigned)((n + 255) / 256)), 256, 0, stream>>>(w, d, O, K, Kpad);
    wcur += (size_t)((n + 31) / 32) * 32;
    return d;
  };

  // ---- consume params in insertion order, fold BN ----
  F f_conv1 = foldCB(getCB(), 4, 64, 7);
  struct Res { F c1, c2, c3, sc; int cin, cmid, cout, stride; bool has_sc; };
  auto mkRes = [&](int cin, int cmid, int cout, int stride, bool sc) -> Res {
    Res r;
    r.c1 = foldCB(getCB(), cin, cmid, 1);
    r.c2 = foldCB(getCB(), cmid, cmid, 3);
    r.c3 = foldCB(getCB(), cmid, cout, 1);
    if (sc) r.sc = foldCB(getCB(), cin, cout, 1); else r.sc = F{nullptr, nullptr, 0};
    r.cin = cin; r.cmid = cmid; r.cout = cout; r.stride = stride; r.has_sc = sc;
    return r;
  };
  Res blk[9] = {
    mkRes(64, 64, 256, 1, true),   mkRes(256, 64, 256, 1, false),  mkRes(256, 64, 256, 1, false),
    mkRes(256, 128, 512, 2, true), mkRes(512, 128, 512, 1, false), mkRes(512, 128, 512, 1, false),
    mkRes(512, 256, 1024, 2, true),mkRes(1024, 256, 1024, 1, false),mkRes(1024, 256, 1024, 1, false)
  };
  F f_c4e = foldCB(getCB(), 1024, 512, 3);
  F f_c4f = foldCB(getCB(), 512, 256, 3);
  const float* fct1w = in_f(); const float* fct1b = in_f();
  const float* fct2w = in_f(); const float* fct2b = in_f();
  const float* fcs1w = in_f(); const float* fcs1b = in_f();
  const float* fcs2w = in_f(); const float* fcs2b = in_f();

  const _Float16* w1t = cvt(fct1w, 256, 65536);
  const _Float16* w2t = cvt(fct2w, 33, 256);
  const _Float16* w1s = cvt(fcs1w, 256, 65536);
  const _Float16* w2s = cvt(fcs2w, 20, 256);

  auto conv = [&](const float* in, int Cin, int H, int W, const F& f, int k,
                  int stride, int pad, int Cout, float* out, const float* resid, int relu) {
    int OH = (H + 2 * pad - k) / stride + 1;
    int OW = (W + 2 * pad - k) / stride + 1;
    int Npos = N * OH * OW;
    dim3 grid((unsigned)((Cout + 127) / 128), (unsigned)((Npos + 15) / 16));
    if (k == 1)
      k_conv<1, 1><<<grid, 256, 0, stream>>>(in, f.w, f.b, resid, out, N, Cin, H, W,
                                             Cout, stride, pad, OH, OW, f.Kpad, relu);
    else if (k == 3)
      k_conv<3, 3><<<grid, 256, 0, stream>>>(in, f.w, f.b, resid, out, N, Cin, H, W,
                                             Cout, stride, pad, OH, OW, f.Kpad, relu);
    else
      k_conv<7, 7><<<grid, 256, 0, stream>>>(in, f.w, f.b, resid, out, N, Cin, H, W,
                                             Cout, stride, pad, OH, OW, f.Kpad, relu);
  };

  // ---- trunk ----
  conv(X, 4, 128, 128, f_conv1, 7, 1, 3, 64, bufA, nullptr, 1);
  { int n = N * 64 * 63 * 63;
    k_maxpool<<<(n + 255) / 256, 256, 0, stream>>>(bufA, bufB, N, 64, 128, 128, 63, 63); }

  float* cur = bufB; float* t1 = bufA; float* t2 = bufC;
  int H = 63, W = 63;
  for (int i = 0; i < 9; ++i) {
    Res& r = blk[i];
    if (r.has_sc) {
      int H2 = (H - 1) / r.stride + 1, W2 = (W - 1) / r.stride + 1;
      conv(cur, r.cin, H, W, r.c1, 1, r.stride, 0, r.cmid, t1, nullptr, 1);
      conv(t1, r.cmid, H2, W2, r.c2, 3, 1, 1, r.cmid, t2, nullptr, 1);
      conv(cur, r.cin, H, W, r.sc, 1, r.stride, 0, r.cout, t1, nullptr, 0);
      conv(t2, r.cmid, H2, W2, r.c3, 1, 1, 0, r.cout, cur, t1, 1);
      H = H2; W = W2;
    } else {
      conv(cur, r.cin, H, W, r.c1, 1, 1, 0, r.cmid, t1, nullptr, 1);
      conv(t1, r.cmid, H, W, r.c2, 3, 1, 1, r.cmid, t2, nullptr, 1);
      conv(t2, r.cmid, H, W, r.c3, 1, 1, 0, r.cout, t1, cur, 1);
      float* tmp = cur; cur = t1; t1 = tmp;
    }
  }
  conv(cur, 1024, 16, 16, f_c4e, 3, 1, 1, 512, t1, nullptr, 1);
  conv(t1, 512, 16, 16, f_c4f, 3, 1, 1, 256, t2, nullptr, 1);

  // ---- flatten to f16 ----
  { long n = (long)N * 65536;
    k_f2h<<<dim3((unsigned)((n + 255) / 256)), 256, 0, stream>>>(t2, XH, n); }

  // ---- FC heads ----
  { int n = N * 256;
    k_zero<<<(n + 255) / 256, 256, 0, stream>>>(HT, n);
    k_zero<<<(n + 255) / 256, 256, 0, stream>>>(HS, n); }
  { int Kpad = 65536, ksplit = 64, kSpan = Kpad / ksplit;
    dim3 g(2, 16, (unsigned)ksplit);
    k_fc<<<g, 32, 0, stream>>>(XH, w1t, nullptr, HT, N, 256, Kpad, kSpan, 0, 0);
    k_fc<<<g, 32, 0, stream>>>(XH, w1s, nullptr, HS, N, 256, Kpad, kSpan, 0, 0); }
  { int n = N * 256;
    k_fc_epi<<<(n + 255) / 256, 256, 0, stream>>>(HT, fct1b, HT16, 256, n);
    k_fc_epi<<<(n + 255) / 256, 256, 0, stream>>>(HS, fcs1b, HS16, 256, n); }

  float* out = (float*)d_out;
  float* pos = out;                 // 32*21*3 = 2016
  float* scl = out + 2016;          // 32*20   = 640
  float* tht = out + 2016 + 640;    // 32*33   = 1056
  { dim3 g(2, 3, 1);
    k_fc<<<g, 32, 0, stream>>>(HT16, w2t, fct2b, tht, N, 33, 256, 256, 1, 3); }
  { dim3 g(2, 2, 1);
    k_fc<<<g, 32, 0, stream>>>(HS16, w2s, fcs2b, scl, N, 20, 256, 256, 1, 2); }

  k_fk<<<1, 32, 0, stream>>>(scl, tht, pos, N);
}
